// MultiHeadAttention_81484119539750
// MI455X (gfx1250) — matrix operand order repaired and checked by reading, not compiled, not run on hardware
//
#include <hip/hip_runtime.h>

// ---------------------------------------------------------------------------
// MultiHeadAttention for MI455X (gfx1250), bf16 WMMA pipeline, fp32 I/O.
//   B=4, T=2048, D=1024, H=16, HD=64
// Stages:
//   1) convert x -> bf16 (row-major)
//   2) transpose-convert W_qkv -> bf16 (3072 x 1024, K-major), W_out likewise
//   3) QKV GEMM, 64x64 tile/wave (16 wmma per K-chunk, 1:1 load:wmma),
//      epilogue scatters Q(*0.125)/K into (B,H,T,HD) and V into (B,H,HD,T)
//   4) causal flash attention, 32 queries/wave, 32-key tiles, online softmax,
//      K/V fragments reused across two query row-blocks
//   5) output projection GEMM -> fp32 d_out
// High-register kernels use __launch_bounds__(N, 1) so the backend allocates
// the full VGPR window instead of spilling accumulators to scratch.
// ---------------------------------------------------------------------------

#define B_  4
#define T_  2048
#define D_  1024
#define H_  16
#define HD_ 64
#define D3_ 3072
#define BT_ (B_ * T_)          // 8192

typedef __attribute__((ext_vector_type(16))) __bf16 v16bf;
typedef __attribute__((ext_vector_type(8)))  float  v8f;
typedef unsigned short u16;
typedef unsigned int   u32;

union Frag16 { v16bf v; uint4 q[2]; };

__device__ __forceinline__ u16 f32_to_bf16(float f) {
    union { float f; u32 u; } x; x.f = f;
    u32 r = x.u + 0x7FFFu + ((x.u >> 16) & 1u);   // round-to-nearest-even
    return (u16)(r >> 16);
}

// A-fragment (16x32 bf16): e<8 at base[e], e>=8 at base[16+e-8]
__device__ __forceinline__ Frag16 load_afrag(const u16* p) {
    Frag16 f;
    f.q[0] = *(const uint4*)(p);
    f.q[1] = *(const uint4*)(p + 16);
    return f;
}
// B-fragment (32x16 bf16): e<8 at base[e], e>=8 at base[16+e-8]
__device__ __forceinline__ Frag16 load_bfrag(const u16* p) {
    Frag16 f;
    f.q[0] = *(const uint4*)(p);
    f.q[1] = *(const uint4*)(p + 16);
    return f;
}

__device__ __forceinline__ v8f wmma_bf16(const Frag16& a, const Frag16& b, v8f c) {
    return __builtin_amdgcn_wmma_f32_16x16x32_bf16(
        /*neg_a=*/false, a.v, /*neg_b=*/false, b.v,
        /*c_mod=*/(short)0, c, /*reuse_a=*/false, /*reuse_b=*/false);
}

// ---------------------------------------------------------------------------
// Stage 1: elementwise fp32 -> bf16
// ---------------------------------------------------------------------------
__global__ void k_convert(const float* __restrict__ in, u16* __restrict__ out, int n) {
    int i = blockIdx.x * blockDim.x + threadIdx.x;
    if (i < n) out[i] = f32_to_bf16(in[i]);
}

// Stage 2: out[c*K + k] = bf16(in[k*C + c])   (make weights K-major)
__global__ void k_transpose_convert(const float* __restrict__ in, u16* __restrict__ out,
                                    int K, int C) {
    int i = blockIdx.x * blockDim.x + threadIdx.x;
    if (i >= K * C) return;
    int c = i / K, k = i % K;
    out[(size_t)c * K + k] = f32_to_bf16(in[(size_t)k * C + c]);
}

// ---------------------------------------------------------------------------
// Stage 3: QKV projection GEMM. One wave computes a 64(M) x 64(N) tile.
// A = x_bf16 (8192 x 1024 row-major), B = W_qkvT (3072 x 1024, K-major).
// ---------------------------------------------------------------------------
__global__ __launch_bounds__(256, 1)
void k_qkv_gemm(const u16* __restrict__ xb, const u16* __restrict__ wt,
                const float* __restrict__ bias,
                u16* __restrict__ qh, u16* __restrict__ kh,
                u16* __restrict__ vt) {
    const int NT = D3_ / 64;                                  // 48
    int wid  = blockIdx.x * (blockDim.x >> 5) + (threadIdx.x >> 5);
    int lane = threadIdx.x & 31;
    int ln   = lane & 15;
    int half = lane >> 4;
    int mt = wid / NT, nt = wid % NT;
    int m0 = mt * 64, n0 = nt * 64;
    int kbA = half * 8, kbB = half * 8;

    v8f acc[4][4];
    v8f z = {0.f,0.f,0.f,0.f,0.f,0.f,0.f,0.f};
#pragma unroll
    for (int i = 0; i < 4; ++i)
#pragma unroll
        for (int j = 0; j < 4; ++j) acc[i][j] = z;

    for (int kk = 0; kk < D_ / 32; ++kk) {
        const u16* ar = xb + (size_t)(m0 + ln) * D_ + kk * 32 + kbA;
        Frag16 a[4];
#pragma unroll
        for (int i = 0; i < 4; ++i) a[i] = load_afrag(ar + i * 16 * D_);
        __builtin_prefetch(ar + 256, 0, 1);                   // 8 chunks ahead
        const u16* brow0 = wt + (size_t)(n0 + ln) * D_ + kk * 32 + kbB;
        __builtin_prefetch(brow0 + 256, 0, 1);
#pragma unroll
        for (int nf = 0; nf < 4; ++nf) {
            Frag16 b = load_bfrag(brow0 + (size_t)(nf * 16) * D_);
#pragma unroll
            for (int i = 0; i < 4; ++i) acc[i][nf] = wmma_bf16(a[i], b, acc[i][nf]);
        }
    }

#pragma unroll
    for (int i = 0; i < 4; ++i)
#pragma unroll
        for (int nf = 0; nf < 4; ++nf) {
            int cidx = n0 + nf * 16 + ln;
            float bv = bias[cidx];
            int h = cidx / (3 * HD_);
            int rem = cidx % (3 * HD_);
            int which = rem / HD_;
            int d = rem % HD_;
#pragma unroll
            for (int r = 0; r < 8; ++r) {
                int tg = m0 + i * 16 + r + half * 8;
                int bb = tg >> 11;            // / T_
                int t  = tg & (T_ - 1);
                float val = acc[i][nf][r] + bv;
                if (which == 0) {
                    qh[((size_t)(bb * H_ + h) * T_ + t) * HD_ + d] =
                        f32_to_bf16(val * 0.125f);        // fold 1/sqrt(64)
                } else if (which == 1) {
                    kh[((size_t)(bb * H_ + h) * T_ + t) * HD_ + d] = f32_to_bf16(val);
                } else {
                    vt[((size_t)(bb * H_ + h) * HD_ + d) * T_ + t] = f32_to_bf16(val);
                }
            }
        }
}

// ---------------------------------------------------------------------------
// Stage 4: causal flash attention. One wave = one (b,h, 32-query tile).
// Per 32-key tile: S = Q K^T (8 wmma), mask, online softmax,
// P routed C-layout -> A-layout through per-wave LDS, O += P V (8 wmma).
// K and V fragments are each reused by both query row-blocks.
// ---------------------------------------------------------------------------
__global__ __launch_bounds__(128, 1)
void k_attention(const u16* __restrict__ qh, const u16* __restrict__ kh,
                 const u16* __restrict__ vt, u16* __restrict__ ahb) {
    __shared__ __align__(16) u16 Plds[4][32 * 32];   // 2KB per wave

    int wl   = threadIdx.x >> 5;
    int lane = threadIdx.x & 31;
    int ln   = lane & 15;
    int half = lane >> 4;
    int kbA = half * 8, kbB = half * 8;

    int wid   = blockIdx.x * 4 + wl;                 // 0 .. B*H*T/32-1
    int qtile = wid % (T_ / 32);
    int bh    = wid / (T_ / 32);
    int q0    = qtile * 32;

    const u16* qp = qh  + (size_t)bh * T_ * HD_;
    const u16* kp = kh  + (size_t)bh * T_ * HD_;
    const u16* vp = vt  + (size_t)bh * HD_ * T_;

    // Q A-fragments: [row block i][dim chunk j], scale already folded in
    Frag16 aq[2][2];
#pragma unroll
    for (int i = 0; i < 2; ++i) {
        const u16* qrow = qp + (size_t)(q0 + i * 16 + ln) * HD_ + kbA;
        aq[i][0] = load_afrag(qrow);
        aq[i][1] = load_afrag(qrow + 32);
    }

    float rm[2][8], rl[2][8];
    v8f o[2][4];
    v8f z = {0.f,0.f,0.f,0.f,0.f,0.f,0.f,0.f};
#pragma unroll
    for (int i = 0; i < 2; ++i)
#pragma unroll
        for (int r = 0; r < 8; ++r) { rm[i][r] = -1e30f; rl[i][r] = 0.f; }
#pragma unroll
    for (int i = 0; i < 2; ++i)
#pragma unroll
        for (int f = 0; f < 4; ++f) o[i][f] = z;

    u16* pl = &Plds[wl][0];
    const int ktmax = qtile;                         // (q0+31)/32

    for (int kt = 0; kt <= ktmax; ++kt) {
        int kbase = kt * 32;
        v8f s[2][2];
#pragma unroll
        for (int i = 0; i < 2; ++i) { s[i][0] = z; s[i][1] = z; }
        // S = Q K^T ; each K fragment pair feeds both row blocks
#pragma unroll
        for (int nf = 0; nf < 2; ++nf) {
            const u16* krow = kp + (size_t)(kbase + nf * 16 + ln) * HD_ + kbB;
            __builtin_prefetch(krow + 32 * HD_, 0, 1);       // next key tile
            Frag16 bk0 = load_bfrag(krow);
            Frag16 bk1 = load_bfrag(krow + 32);
#pragma unroll
            for (int i = 0; i < 2; ++i) {
                s[i][nf] = wmma_bf16(aq[i][0], bk0, s[i][nf]);
                s[i][nf] = wmma_bf16(aq[i][1], bk1, s[i][nf]);
            }
        }
        // causal mask on the diagonal tile
        if (kbase + 31 > q0) {
#pragma unroll
            for (int i = 0; i < 2; ++i)
#pragma unroll
                for (int nf = 0; nf < 2; ++nf) {
                    int key = kbase + nf * 16 + ln;
#pragma unroll
                    for (int r = 0; r < 8; ++r) {
                        int qrow_i = q0 + i * 16 + r + half * 8;
                        if (key > qrow_i) s[i][nf][r] = -1e30f;
                    }
                }
        }
        // online softmax (row stats replicated across the 16-lane half-group)
#pragma unroll
        for (int i = 0; i < 2; ++i)
#pragma unroll
            for (int r = 0; r < 8; ++r) {
                float v = fmaxf(s[i][0][r], s[i][1][r]);
                v = fmaxf(v, __shfl_xor(v, 1));
                v = fmaxf(v, __shfl_xor(v, 2));
                v = fmaxf(v, __shfl_xor(v, 4));
                v = fmaxf(v, __shfl_xor(v, 8));
                float nm = fmaxf(rm[i][r], v);
                float sc = __expf(rm[i][r] - nm);
                rm[i][r] = nm;
                float p0 = __expf(s[i][0][r] - nm);
                float p1 = __expf(s[i][1][r] - nm);
                s[i][0][r] = p0; s[i][1][r] = p1;
                float ts = p0 + p1;
                ts += __shfl_xor(ts, 1);
                ts += __shfl_xor(ts, 2);
                ts += __shfl_xor(ts, 4);
                ts += __shfl_xor(ts, 8);
                rl[i][r] = rl[i][r] * sc + ts;
#pragma unroll
                for (int f = 0; f < 4; ++f) o[i][f][r] *= sc;
            }
        // P: C-layout -> row-major LDS (32 rows x 32 keys, bf16)
#pragma unroll
        for (int i = 0; i < 2; ++i)
#pragma unroll
            for (int r = 0; r < 8; ++r) {
                int m = i * 16 + r + half * 8;
                pl[m * 32 + ln]      = f32_to_bf16(s[i][0][r]);
                pl[m * 32 + 16 + ln] = f32_to_bf16(s[i][1][r]);
            }
        asm volatile("s_wait_dscnt 0" ::: "memory");
        Frag16 ap[2];
        ap[0] = load_afrag(pl + (size_t)ln * 32 + kbA);
        ap[1] = load_afrag(pl + (size_t)(16 + ln) * 32 + kbA);
        // O += P * V ; each V fragment feeds both row blocks
#pragma unroll
        for (int f = 0; f < 4; ++f) {
            const u16* vrow = vp + (size_t)(f * 16 + ln) * T_ + kbase + kbB;
            __builtin_prefetch(vrow + 32, 0, 1);             // next key tile
            Frag16 bv = load_bfrag(vrow);
            o[0][f] = wmma_bf16(ap[0], bv, o[0][f]);
            o[1][f] = wmma_bf16(ap[1], bv, o[1][f]);
        }
    }

    // epilogue: normalize, write bf16 attention output in (B,T,D)
    int bb = bh / H_, h = bh % H_;
#pragma unroll
    for (int i = 0; i < 2; ++i)
#pragma unroll
        for (int f = 0; f < 4; ++f) {
            int dcol = h * HD_ + f * 16 + ln;
#pragma unroll
            for (int r = 0; r < 8; ++r) {
                int t = q0 + i * 16 + r + half * 8;
                ahb[((size_t)bb * T_ + t) * D_ + dcol] = f32_to_bf16(o[i][f][r] / rl[i][r]);
            }
        }
}

// ---------------------------------------------------------------------------
// Stage 5: output projection GEMM -> fp32. 64x64 tile per wave.
// ---------------------------------------------------------------------------
__global__ __launch_bounds__(256, 1)
void k_out_gemm(const u16* __restrict__ ab, const u16* __restrict__ wt,
                const float* __restrict__ bias, float* __restrict__ out) {
    const int NT = D_ / 64;                                   // 16
    int wid  = blockIdx.x * (blockDim.x >> 5) + (threadIdx.x >> 5);
    int lane = threadIdx.x & 31;
    int ln   = lane & 15;
    int half = lane >> 4;
    int mt = wid / NT, nt = wid % NT;
    int m0 = mt * 64, n0 = nt * 64;
    int kbA = half * 8, kbB = half * 8;

    v8f acc[4][4];
    v8f z = {0.f,0.f,0.f,0.f,0.f,0.f,0.f,0.f};
#pragma unroll
    for (int i = 0; i < 4; ++i)
#pragma unroll
        for (int j = 0; j < 4; ++j) acc[i][j] = z;

    for (int kk = 0; kk < D_ / 32; ++kk) {
        const u16* ar = ab + (size_t)(m0 + ln) * D_ + kk * 32 + kbA;
        Frag16 a[4];
#pragma unroll
        for (int i = 0; i < 4; ++i) a[i] = load_afrag(ar + i * 16 * D_);
        __builtin_prefetch(ar + 256, 0, 1);
        const u16* brow0 = wt + (size_t)(n0 + ln) * D_ + kk * 32 + kbB;
        __builtin_prefetch(brow0 + 256, 0, 1);
#pragma unroll
        for (int nf = 0; nf < 4; ++nf) {
            Frag16 b = load_bfrag(brow0 + (size_t)(nf * 16) * D_);
#pragma unroll
            for (int i = 0; i < 4; ++i) acc[i][nf] = wmma_bf16(a[i], b, acc[i][nf]);
        }
    }

#pragma unroll
    for (int i = 0; i < 4; ++i)
#pragma unroll
        for (int nf = 0; nf < 4; ++nf) {
            int cidx = n0 + nf * 16 + ln;
            float bv = bias[cidx];
#pragma unroll
            for (int r = 0; r < 8; ++r) {
                int tg = m0 + i * 16 + r + half * 8;
                out[(size_t)tg * D_ + cidx] = acc[i][nf][r] + bv;
            }
        }
}

// ---------------------------------------------------------------------------
// Launch
// ---------------------------------------------------------------------------
extern "C" void kernel_launch(void* const* d_in, const int* in_sizes, int n_in,
                              void* d_out, int out_size, void* d_ws, size_t ws_size,
                              hipStream_t stream) {
    const float* x     = (const float*)d_in[0];
    const float* Wqkv  = (const float*)d_in[1];
    const float* bqkv  = (const float*)d_in[2];
    const float* Wout  = (const float*)d_in[3];
    const float* bout  = (const float*)d_in[4];
    float* out = (float*)d_out;

    char* ws = (char*)d_ws;
    const size_t SZ_XB   = (size_t)BT_ * D_ * 2;       // 16 MB
    const size_t SZ_WQT  = (size_t)D_ * D3_ * 2;       // 6 MB
    const size_t SZ_WOT  = (size_t)D_ * D_ * 2;        // 2 MB
    const size_t SZ_HEAD = (size_t)B_ * H_ * T_ * HD_ * 2;  // 16 MB

    u16* xb   = (u16*)(ws);
    u16* wqt  = (u16*)(ws + SZ_XB);
    u16* wot  = (u16*)(ws + SZ_XB + SZ_WQT);
    u16* qh   = (u16*)(ws + SZ_XB + SZ_WQT + SZ_WOT);
    u16* kh   = (u16*)(ws + SZ_XB + SZ_WQT + SZ_WOT + SZ_HEAD);
    u16* vt   = (u16*)(ws + SZ_XB + SZ_WQT + SZ_WOT + 2 * SZ_HEAD);
    u16* ahb  = (u16*)(ws + SZ_XB + SZ_WQT + SZ_WOT + 3 * SZ_HEAD);

    // Stage 1/2: conversions
    {
        int n = BT_ * D_;
        k_convert<<<dim3((n + 255) / 256), dim3(256), 0, stream>>>(x, xb, n);
    }
    {
        int n = D_ * D3_;
        k_transpose_convert<<<dim3((n + 255) / 256), dim3(256), 0, stream>>>(Wqkv, wqt, D_, D3_);
    }
    {
        int n = D_ * D_;
        k_transpose_convert<<<dim3((n + 255) / 256), dim3(256), 0, stream>>>(Wout, wot, D_, D_);
    }
    // Stage 3: QKV GEMM: (8192/64)*(3072/64) = 6144 wave-tiles, 8 waves/block
    k_qkv_gemm<<<dim3(6144 / 8), dim3(256), 0, stream>>>(xb, wqt, bqkv, qh, kh, vt);
    // Stage 4: attention: B*H*(T/32) = 4096 wave-tiles, 4 waves/block
    k_attention<<<dim3(4096 / 4), dim3(128), 0, stream>>>(qh, kh, vt, ahb);
    // Stage 5: output GEMM: (8192/64)*(1024/64) = 2048 wave-tiles, 8 waves/block
    k_out_gemm<<<dim3(2048 / 8), dim3(256), 0, stream>>>(ahb, wot, bout, out);
}
